// GNNWrapper_18090402251415
// MI455X (gfx1250) — compile-verified, matrix operand order checked by reading, not run on hardware
//
#include <hip/hip_runtime.h>
#include <math.h>

// MI455X / gfx1250, wave32. Fused ECCConv via v_wmma_f32_16x16x32_bf16.
// B=1024, N=16, F=4, S=4, C=128, H=256, U=256

typedef __attribute__((ext_vector_type(16))) __bf16 v16bf;
typedef __attribute__((ext_vector_type(8)))  __bf16 v8bf;
typedef __attribute__((ext_vector_type(8)))  float  v8f;

#define NB   1024
#define NN   16
#define NF   4
#define NS   4
#define NC   128
#define NH   256
#define NU   256

// K index inside a 16-bit WMMA A/B fragment: element e (0..15), lane-half h.
// VGPR v=e/2 pos=e&1; v<4 -> K = 2v+pos+8h ; v>=4 -> K = 16 + 2(v-4)+pos+8h.
__device__ __forceinline__ int frag_k(int e, int half) {
  int v = e >> 1, pos = e & 1;
  return ((v & 4) << 2) + ((v & 3) << 1) + pos + (half << 3);
}

__device__ __forceinline__ v8f wmma_bf16(v16bf a, v16bf b, v8f c) {
  return __builtin_amdgcn_wmma_f32_16x16x32_bf16(
      /*neg_a=*/false, a, /*neg_b=*/false, b,
      /*c_mod=*/(short)0, c, /*reuse_a=*/false, /*reuse_b=*/false);
}

// Load a 16-row A fragment (rows = lm, K window k0..k0+31) from an LDS
// [16][256] bf16 tile. Two 16-byte ds loads per lane.
__device__ __forceinline__ v16bf load_afrag(const __bf16* row_k0, int half) {
  v8bf lo = *(const v8bf*)(row_k0 + 8 * half);        // e0..7  -> K 8h..8h+7
  v8bf hi = *(const v8bf*)(row_k0 + 16 + 8 * half);   // e8..15 -> K 16+8h..
  return __builtin_shufflevector(lo, hi, 0, 1, 2, 3, 4, 5, 6, 7,
                                 8, 9, 10, 11, 12, 13, 14, 15);
}

// ---------------------------------------------------------------------------
// Prep: pack W1 (zero-padded K 4->32), W2, W3 into bf16 B-fragment order.
// Fragment = 32 lanes x 16 bf16 (512 elems, 1KB). idx = lane*16 + e.
// fW1: 16 frags | fW2: 8*16 frags | fW3: 8*32 frags.
// ---------------------------------------------------------------------------
__global__ __launch_bounds__(512) void ecc_prep_frags(
    const float* __restrict__ W1, const float* __restrict__ W2,
    const float* __restrict__ W3, __bf16* __restrict__ fW1,
    __bf16* __restrict__ fW2, __bf16* __restrict__ fW3) {
  int fid = blockIdx.x;            // 0..399
  int idx = threadIdx.x;           // 0..511
  int lane = idx >> 4, e = idx & 15;
  int half = lane >> 4, lm = lane & 15;
  int kk = frag_k(e, half);        // 0..31
  if (fid < 16) {                  // W1 [4,256], K padded with zeros
    int n = fid * 16 + lm;
    float v = (kk < NS) ? W1[kk * NH + n] : 0.0f;
    fW1[fid * 512 + idx] = (__bf16)v;
  } else if (fid < 144) {          // W2 [256,256]
    int f2 = fid - 16;
    int ks = f2 >> 4, nt = f2 & 15;
    int n = nt * 16 + lm, K = ks * 32 + kk;
    fW2[f2 * 512 + idx] = (__bf16)W2[K * NH + n];
  } else {                         // W3 [256,512]
    int f3 = fid - 144;
    int ks = f3 >> 5, nt = f3 & 31;
    int n = nt * 16 + lm, K = ks * 32 + kk;
    fW3[f3 * 512 + idx] = (__bf16)W3[K * (NC * NF) + n];
  }
}

// ---------------------------------------------------------------------------
// Fused ECC conv: one wave = one (b,t) tile of 16 edge rows (s=0..15).
// GEMM1 (E@W1,K=4 padded), GEMM2 (h1@W2), GEMM3 (h2@W3) on bf16 WMMA,
// then accumulator-space contraction with A[b,t,s]*X[b,s,f], root term,
// relu -> Y[b,t,c].
// ---------------------------------------------------------------------------
__global__ __launch_bounds__(256) void ecc_fused(
    const float* __restrict__ X, const float* __restrict__ A,
    const float* __restrict__ E, const float* __restrict__ b1,
    const float* __restrict__ b2, const float* __restrict__ b3,
    const float* __restrict__ root, const float* __restrict__ bias,
    const __bf16* __restrict__ fW1, const __bf16* __restrict__ fW2,
    const __bf16* __restrict__ fW3, float* __restrict__ Y) {
  __shared__ __attribute__((aligned(16))) __bf16 hbuf[8][16 * NH];  // 64KB
  __shared__ float axs[8][64];                                      // 2KB

  const int tid = threadIdx.x;
  const int wave = tid >> 5;
  const int lane = tid & 31;
  const int half = lane >> 4;
  const int lm = lane & 15;
  const int tile = blockIdx.x * 8 + wave;  // 0..16383 == b*16 + t
  const int b = tile >> 4;

  __bf16* H = hbuf[wave];
  float* ax = axs[wave];

  // --- precompute ax[s][f] = A[b,t,s] * X[b,s,f] ---
  if (lane < 16) {
    int s = lane;
    float a = A[(size_t)tile * NN + s];
    const float* Xs = X + ((size_t)b * NN + s) * NF;
#pragma unroll
    for (int f = 0; f < NF; ++f) ax[s * 4 + f] = a * Xs[f];
  }

  // --- GEMM1: h1 = relu(E @ W1 + b1), K=4 zero-padded to 32 ---
  v16bf aE = {};
  {
    const float* Erow = E + (size_t)tile * (NN * NS);
    if (half == 0) {
      float4 e4 = *(const float4*)(Erow + lm * NS);
      aE[0] = (__bf16)e4.x; aE[1] = (__bf16)e4.y;
      aE[2] = (__bf16)e4.z; aE[3] = (__bf16)e4.w;
    }
  }
  for (int nt = 0; nt < 16; ++nt) {
    v16bf bw = ((const v16bf*)(fW1 + nt * 512))[lane];
    v8f acc = {};
    acc = wmma_bf16(aE, bw, acc);
    int j = nt * 16 + lm;
    float bb = b1[j];
#pragma unroll
    for (int v = 0; v < 8; ++v) {
      float hv = fmaxf(acc[v] + bb, 0.0f);
      H[(v + 8 * half) * NH + j] = (__bf16)hv;
    }
  }

  // --- pull h1 A-fragments into registers ---
  v16bf a1[8];
#pragma unroll
  for (int k = 0; k < 8; ++k) a1[k] = load_afrag(H + lm * NH + k * 32, half);

  // --- GEMM2: h2 = relu(h1 @ W2 + b2) ---  (reuse H buffer for h2)
  for (int nt = 0; nt < 16; ++nt) {
    v8f acc = {};
#pragma unroll
    for (int k = 0; k < 8; ++k) {
      v16bf bw = ((const v16bf*)(fW2 + (k * 16 + nt) * 512))[lane];
      acc = wmma_bf16(a1[k], bw, acc);
    }
    int j = nt * 16 + lm;
    float bb = b2[j];
#pragma unroll
    for (int v = 0; v < 8; ++v) {
      float hv = fmaxf(acc[v] + bb, 0.0f);
      H[(v + 8 * half) * NH + j] = (__bf16)hv;
    }
  }

  // --- pull h2 A-fragments ---
  v16bf a2[8];
#pragma unroll
  for (int k = 0; k < 8; ++k) a2[k] = load_afrag(H + lm * NH + k * 32, half);

  // --- per-lane contraction weights: f is fixed per lane (col & 3) ---
  const int fl = lm & 3;
  float axr[8];
#pragma unroll
  for (int v = 0; v < 8; ++v) axr[v] = ax[(v + 8 * half) * 4 + fl];

  const float* Xt = X + (size_t)tile * NF;  // X[b,t,:]
  float xt0 = Xt[0], xt1 = Xt[1], xt2 = Xt[2], xt3 = Xt[3];

  // --- GEMM3 + masked message aggregation + root + relu ---
  for (int nt = 0; nt < 32; ++nt) {
    v8f acc = {};
#pragma unroll
    for (int k = 0; k < 8; ++k) {
      v16bf bw = ((const v16bf*)(fW3 + (k * 32 + nt) * 512))[lane];
      acc = wmma_bf16(a2[k], bw, acc);
    }
    int col = nt * 16 + lm;       // column of k matrix: 4c+f
    float b3c = b3[col];
    float p = 0.0f;
#pragma unroll
    for (int v = 0; v < 8; ++v) p += (acc[v] + b3c) * axr[v];
    // reduce over f (xor 1,2) and the two row-halves (xor 16)
    p += __shfl_xor(p, 1, 32);
    p += __shfl_xor(p, 2, 32);
    p += __shfl_xor(p, 16, 32);
    if (half == 0 && (lm & 3) == 0) {
      int c = nt * 4 + (lm >> 2);
      float r = bias[c] + xt0 * root[c] + xt1 * root[NC + c] +
                xt2 * root[2 * NC + c] + xt3 * root[3 * NC + c];
      Y[(size_t)tile * NC + c] = fmaxf(p + r, 0.0f);
    }
  }
}

// ---------------------------------------------------------------------------
// GlobalAttnSumPool + Dense(tanh). One block per batch element.
// ---------------------------------------------------------------------------
__global__ __launch_bounds__(256) void ecc_pool_dense(
    const float* __restrict__ Y, const float* __restrict__ attn,
    const float* __restrict__ Wd, const float* __restrict__ bd,
    float* __restrict__ out) {
  __shared__ float ly[NN * NC];
  __shared__ float logits[NN];
  __shared__ float aw[NN];
  __shared__ float pooled[NC];
  const int b = blockIdx.x, tid = threadIdx.x;
  const float* Yb = Y + (size_t)b * (NN * NC);
  for (int i = tid; i < NN * NC; i += 256) ly[i] = Yb[i];
  __syncthreads();
  if (tid < NN) {
    float s = 0.0f;
    for (int c = 0; c < NC; ++c) s += ly[tid * NC + c] * attn[c];
    logits[tid] = s;
  }
  __syncthreads();
  if (tid == 0) {
    float m = logits[0];
    for (int n = 1; n < NN; ++n) m = fmaxf(m, logits[n]);
    float w[NN], sum = 0.0f;
    for (int n = 0; n < NN; ++n) { w[n] = __expf(logits[n] - m); sum += w[n]; }
    float inv = 1.0f / sum;
    for (int n = 0; n < NN; ++n) aw[n] = w[n] * inv;
  }
  __syncthreads();
  if (tid < NC) {
    float s = 0.0f;
    for (int n = 0; n < NN; ++n) s += aw[n] * ly[n * NC + tid];
    pooled[tid] = s;
  }
  __syncthreads();
  {
    int u = tid;  // 256 threads == U
    float s = bd[u];
    for (int c = 0; c < NC; ++c) s += pooled[c] * Wd[c * NU + u];
    out[(size_t)b * NU + u] = tanhf(s);
  }
}

// ---------------------------------------------------------------------------
extern "C" void kernel_launch(void* const* d_in, const int* in_sizes, int n_in,
                              void* d_out, int out_size, void* d_ws,
                              size_t ws_size, hipStream_t stream) {
  const float* X    = (const float*)d_in[0];
  const float* A    = (const float*)d_in[1];
  const float* E    = (const float*)d_in[2];
  const float* W1   = (const float*)d_in[3];
  const float* b1   = (const float*)d_in[4];
  const float* W2   = (const float*)d_in[5];
  const float* b2   = (const float*)d_in[6];
  const float* W3   = (const float*)d_in[7];
  const float* b3   = (const float*)d_in[8];
  const float* root = (const float*)d_in[9];
  const float* bias = (const float*)d_in[10];
  const float* attn = (const float*)d_in[11];
  const float* Wd   = (const float*)d_in[12];
  const float* bd   = (const float*)d_in[13];

  char* ws = (char*)d_ws;
  float*  Y   = (float*)ws;                               // 16384*128*4 = 8 MiB
  size_t  off = (size_t)NB * NN * NC * sizeof(float);
  __bf16* fW1 = (__bf16*)(ws + off);  off += 16  * 512 * sizeof(__bf16);
  __bf16* fW2 = (__bf16*)(ws + off);  off += 128 * 512 * sizeof(__bf16);
  __bf16* fW3 = (__bf16*)(ws + off);

  ecc_prep_frags<<<400, 512, 0, stream>>>(W1, W2, W3, fW1, fW2, fW3);
  ecc_fused<<<(NB * NN) / 8, 256, 0, stream>>>(X, A, E, b1, b2, b3, root, bias,
                                               fW1, fW2, fW3, Y);
  ecc_pool_dense<<<NB, 256, 0, stream>>>(Y, attn, Wd, bd, (float*)d_out);
}
